// MultiheadAttention_274877907061
// MI455X (gfx1250) — compile-verified, matrix operand order checked by reading, not run on hardware
//
#include <hip/hip_runtime.h>

// ---------------------------------------------------------------------------
// MI455X multi-head attention, bf16 WMMA (f32 accumulate) throughout.
//   q/k/v = x @ W^T + b     (NT GEMM, both operands K-contiguous row-major)
//   flash-style attention over 128-key tiles with parallel online softmax
//   out = ctx @ W_out^T + b (f32 result)
// LDS staging uses gfx1250 GLOBAL_LOAD_ASYNC_TO_LDS_B128 (ASYNCcnt) when the
// toolchain declares the builtins; otherwise a synchronous fallback.
// ---------------------------------------------------------------------------

typedef __attribute__((ext_vector_type(16))) __bf16 v16bf;
typedef __attribute__((ext_vector_type(8)))  float  v8f;
typedef __attribute__((ext_vector_type(4)))  int    v4i_;

union BF16x16 { v16bf v; unsigned int u[8]; };

struct __align__(16) B128 { unsigned int x[4]; };

#if defined(__AMDGCN__) && __has_builtin(__builtin_amdgcn_global_load_async_to_lds_b128) && \
    __has_builtin(__builtin_amdgcn_s_wait_asynccnt)
#define USE_ASYNC_LDS 1
#else
#define USE_ASYNC_LDS 0
#endif

#if USE_ASYNC_LDS
typedef __attribute__((address_space(1))) v4i_ GV4;  // global int4
typedef __attribute__((address_space(3))) v4i_ LV4;  // LDS int4
#endif

__device__ __forceinline__ void async_cp_b128(const void* gsrc, void* ldst) {
#if USE_ASYNC_LDS
  __builtin_amdgcn_global_load_async_to_lds_b128(
      (GV4*)(void*)gsrc, (LV4*)ldst, 0, 0);
#else
  *(B128*)ldst = *(const B128*)gsrc;
#endif
}

__device__ __forceinline__ void async_wait() {
#if USE_ASYNC_LDS
  __builtin_amdgcn_s_wait_asynccnt(0);
#endif
}

// Load a 16x32 bf16 A/B fragment from an LDS tile stored row-major with
// K contiguous.  Matches the ISA 16-bit 16x32 layout:
//   lanes 0-15 : row = lane,     K = 0..7  (VGPR0..3), K = 16..23 (VGPR4..7)
//   lanes 16-31: row = lane-16,  K = 8..15 (VGPR0..3), K = 24..31 (VGPR4..7)
__device__ __forceinline__ v16bf ld_frag(const __bf16* tile, int row0, int k0, int ld) {
  const int lane = threadIdx.x & 31;
  const int r  = row0 + (lane & 15);
  const int kb = (lane & 16) ? 4 : 0;  // dword base (kbase/2)
  const unsigned int* p = reinterpret_cast<const unsigned int*>(tile + r * ld + k0);
  BF16x16 f;
#pragma unroll
  for (int j = 0; j < 4; ++j) f.u[j]     = p[kb + j];
#pragma unroll
  for (int j = 0; j < 4; ++j) f.u[4 + j] = p[kb + 8 + j];
  return f.v;
}

// ---------------------------------------------------------------------------
__global__ void cvt_f32_bf16(const float* __restrict__ in, __bf16* __restrict__ out, int n) {
  int i = blockIdx.x * blockDim.x + threadIdx.x;
  int stride = gridDim.x * blockDim.x;
  for (; i < n; i += stride) out[i] = (__bf16)in[i];
}

// ---------------------------------------------------------------------------
// C[M,N] = A[M,K](bf16) * W[N,K]^T (bf16) + bias[N], OutT = float or __bf16.
// Block tile 128x128, 256 threads = 8 waves; wave tile 64x32 (4x2 WMMA tiles).
template <typename OutT>
__global__ __launch_bounds__(256)
void gemm_nt_wmma(const __bf16* __restrict__ A, const __bf16* __restrict__ W,
                  const float* __restrict__ bias, OutT* __restrict__ C,
                  int M, int N, int K) {
  __shared__ __align__(16) __bf16 As[128 * 32];
  __shared__ __align__(16) __bf16 Bs[128 * 32];
  const int t    = threadIdx.x;
  const int lane = t & 31;
  const int w    = t >> 5;
  const int wm   = (w >> 2) * 64;
  const int wn   = (w & 3) * 32;
  const int bm   = blockIdx.x * 128;
  const int bn   = blockIdx.y * 128;

  v8f acc[4][2] = {};

  const size_t rowB = (size_t)K * 2;  // bytes per global row
  const char* Ab = (const char*)A + (size_t)bm * rowB;
  const char* Wb = (const char*)W + (size_t)bn * rowB;

  for (int k0 = 0; k0 < K; k0 += 32) {
    const char* Ac = Ab + (size_t)k0 * 2;
    const char* Wc = Wb + (size_t)k0 * 2;
    // 128x32 bf16 tile = 512 x 16B chunks; 4 chunks per row
#pragma unroll
    for (int i = 0; i < 2; ++i) {
      int e = t + i * 256;
      int r = e >> 2, c = e & 3;
      async_cp_b128(Ac + (size_t)r * rowB + c * 16, (char*)As + e * 16);
      async_cp_b128(Wc + (size_t)r * rowB + c * 16, (char*)Bs + e * 16);
    }
    async_wait();
    __syncthreads();

    v16bf af[4], bfr[2];
#pragma unroll
    for (int i = 0; i < 4; ++i) af[i]  = ld_frag(As, wm + i * 16, 0, 32);
#pragma unroll
    for (int j = 0; j < 2; ++j) bfr[j] = ld_frag(Bs, wn + j * 16, 0, 32);
#pragma unroll
    for (int i = 0; i < 4; ++i)
#pragma unroll
      for (int j = 0; j < 2; ++j)
        acc[i][j] = __builtin_amdgcn_wmma_f32_16x16x32_bf16(
            false, af[i], false, bfr[j], (short)0, acc[i][j], false, false);
    __syncthreads();
  }

  // Epilogue: C layout — VGPR v holds M = v (lanes 0-15) / v+8 (lanes 16-31),
  // N = lane & 15.
  const int rhi = (lane >> 4) << 3;
  const int cl  = lane & 15;
#pragma unroll
  for (int i = 0; i < 4; ++i)
#pragma unroll
    for (int j = 0; j < 2; ++j) {
      int col = bn + wn + j * 16 + cl;
      float bb = bias[col];
#pragma unroll
      for (int v = 0; v < 8; ++v) {
        int row = bm + wm + i * 16 + v + rhi;
        C[(size_t)row * N + col] = (OutT)(acc[i][j][v] + bb);
      }
    }
}

// ---------------------------------------------------------------------------
// Flash attention: one block per (b, h, 64-row Q tile), 8 waves, 128-key tiles.
// Scores: wave computes 16(q) x 64(key) of the 64x128 score tile (1x4 WMMA).
// Output: wave accumulates 16(q) x 32(d) of the 64x64 O tile (1x2 WMMA).
__global__ __launch_bounds__(256)
void flash_attn_wmma(const __bf16* __restrict__ Qg, const __bf16* __restrict__ Kg,
                     const __bf16* __restrict__ Vg, __bf16* __restrict__ Og,
                     int S_, int D_, int H_) {
  __shared__ __align__(16) __bf16 Qt[64 * 64];    // [q][d]
  __shared__ __align__(16) __bf16 Kt[128 * 64];   // [key][d]
  __shared__ __align__(16) __bf16 Vt[64 * 128];   // [d][key] (transposed)
  __shared__ __align__(16) __bf16 Pt[64 * 128];   // exp'd probs, bf16
  __shared__ float Sb[64 * 128];                  // scaled scores
  __shared__ float pmax[64 * 4], psum[64 * 4];
  __shared__ float mrow[64], lrow[64], crow[64];

  const int t = threadIdx.x, lane = t & 31, w = t >> 5;
  const int qt_ = (w >> 1) * 16;   // wave's q-row offset
  const int cs  = (w & 1) * 64;    // wave's score-col offset (128-wide tile)
  const int co  = (w & 1) * 32;    // wave's output-col offset (64-wide tile)
  const int bh = blockIdx.y;
  const int b = bh / H_, h = bh % H_;
  const int q0 = blockIdx.x * 64;
  const size_t rowB = (size_t)D_ * 2;  // bytes per global row

  if (t < 64) { mrow[t] = -1e30f; lrow[t] = 0.0f; }

  // Stage Q tile: 64x64 bf16 = 512 x 16B chunks, 8 per row
  const char* Qc = (const char*)(Qg + ((size_t)(b * S_ + q0)) * D_ + (size_t)h * 64);
#pragma unroll
  for (int i = 0; i < 2; ++i) {
    int e = t + i * 256;
    int r = e >> 3, c = e & 7;
    async_cp_b128(Qc + (size_t)r * rowB + c * 16, (char*)Qt + e * 16);
  }

  v8f acc_o[2] = {};
  const float scale = 0.125f;  // 1/sqrt(DH=64)

  for (int key0 = 0; key0 < S_; key0 += 128) {
    // --- stage K tile [key][d]: 128x64 bf16 = 1024 x 16B chunks
    const char* Kc = (const char*)(Kg + ((size_t)(b * S_ + key0)) * D_ + (size_t)h * 64);
#pragma unroll
    for (int i = 0; i < 4; ++i) {
      int e = t + i * 256;
      int r = e >> 3, c = e & 7;
      async_cp_b128(Kc + (size_t)r * rowB + c * 16, (char*)Kt + e * 16);
    }
    // --- stage V tile transposed [d][key]
    const __bf16* Vs = Vg + ((size_t)(b * S_ + key0)) * D_ + (size_t)h * 64;
#pragma unroll
    for (int i = 0; i < 32; ++i) {
      int e = t + i * 256;            // 8192 elements
      int kk = e >> 6, dd = e & 63;
      Vt[dd * 128 + kk] = Vs[(size_t)kk * D_ + dd];
    }
    async_wait();
    __syncthreads();

    // --- scores = Q * K^T (K-dim = DH = 64, two WMMA K-steps, 4 col tiles)
    v8f acc_s[4] = {};
#pragma unroll
    for (int d0 = 0; d0 < 64; d0 += 32) {
      v16bf a = ld_frag(Qt, qt_, d0, 64);
#pragma unroll
      for (int j = 0; j < 4; ++j) {
        v16bf bk = ld_frag(Kt, cs + j * 16, d0, 64);
        acc_s[j] = __builtin_amdgcn_wmma_f32_16x16x32_bf16(
            false, a, false, bk, (short)0, acc_s[j], false, false);
      }
    }
    {
      const int rhi = (lane >> 4) << 3, cl = lane & 15;
#pragma unroll
      for (int j = 0; j < 4; ++j)
#pragma unroll
        for (int v = 0; v < 8; ++v)
          Sb[(qt_ + v + rhi) * 128 + cs + j * 16 + cl] = acc_s[j][v] * scale;
    }
    __syncthreads();

    // --- parallel online softmax: 4 threads per row, 32 cols each
    {
      const int r = t >> 2, seg = t & 3;
      const float* sr = &Sb[r * 128 + seg * 32];
      float mx = -1e30f;
#pragma unroll
      for (int c = 0; c < 32; ++c) mx = fmaxf(mx, sr[c]);
      pmax[r * 4 + seg] = mx;
    }
    __syncthreads();
    {
      const int r = t >> 2, seg = t & 3;
      float newm = fmaxf(fmaxf(pmax[r * 4 + 0], pmax[r * 4 + 1]),
                         fmaxf(pmax[r * 4 + 2], pmax[r * 4 + 3]));
      newm = fmaxf(newm, mrow[r]);
      const float* sr = &Sb[r * 128 + seg * 32];
      __bf16* pr = &Pt[r * 128 + seg * 32];
      float s = 0.0f;
#pragma unroll
      for (int c = 0; c < 32; ++c) {
        float pv = __expf(sr[c] - newm);
        pr[c] = (__bf16)pv;
        s += pv;
      }
      psum[r * 4 + seg] = s;
    }
    __syncthreads();
    if (t < 64) {
      float newm = fmaxf(fmaxf(pmax[t * 4 + 0], pmax[t * 4 + 1]),
                         fmaxf(pmax[t * 4 + 2], pmax[t * 4 + 3]));
      newm = fmaxf(newm, mrow[t]);
      float corr = __expf(mrow[t] - newm);
      lrow[t] = lrow[t] * corr +
                (psum[t * 4 + 0] + psum[t * 4 + 1] + psum[t * 4 + 2] + psum[t * 4 + 3]);
      mrow[t] = newm;
      crow[t] = corr;
    }
    __syncthreads();

    // --- rescale running output, then O += P * V (K-dim = 128 keys)
    {
      const int rhi = (lane >> 4) << 3;
#pragma unroll
      for (int j = 0; j < 2; ++j)
#pragma unroll
        for (int v = 0; v < 8; ++v)
          acc_o[j][v] *= crow[qt_ + v + rhi];
    }
#pragma unroll
    for (int k0 = 0; k0 < 128; k0 += 32) {
      v16bf a = ld_frag(Pt, qt_, k0, 128);
#pragma unroll
      for (int j = 0; j < 2; ++j) {
        v16bf bv = ld_frag(Vt, co + j * 16, k0, 128);
        acc_o[j] = __builtin_amdgcn_wmma_f32_16x16x32_bf16(
            false, a, false, bv, (short)0, acc_o[j], false, false);
      }
    }
    __syncthreads();
  }

  if (t < 64) crow[t] = 1.0f / lrow[t];
  __syncthreads();

  const int rhi = (lane >> 4) << 3, cl = lane & 15;
#pragma unroll
  for (int j = 0; j < 2; ++j)
#pragma unroll
    for (int v = 0; v < 8; ++v) {
      int row = qt_ + v + rhi;
      int col = co + j * 16 + cl;
      Og[((size_t)(b * S_ + q0 + row)) * D_ + (size_t)h * 64 + col] =
          (__bf16)(acc_o[j][v] * crow[row]);
    }
}

// ---------------------------------------------------------------------------
extern "C" void kernel_launch(void* const* d_in, const int* in_sizes, int n_in,
                              void* d_out, int out_size, void* d_ws, size_t ws_size,
                              hipStream_t stream) {
  (void)in_sizes; (void)n_in; (void)out_size; (void)ws_size;
  const int B = 4, S = 2048, D = 1024, H = 16;
  const float* x_q  = (const float*)d_in[0];
  const float* x_kv = (const float*)d_in[1];
  const float* W_q  = (const float*)d_in[2];
  const float* b_q  = (const float*)d_in[3];
  const float* W_k  = (const float*)d_in[4];
  const float* b_k  = (const float*)d_in[5];
  const float* W_v  = (const float*)d_in[6];
  const float* b_v  = (const float*)d_in[7];
  const float* W_o  = (const float*)d_in[8];
  const float* b_o  = (const float*)d_in[9];
  float* out = (float*)d_out;

  const size_t MS = (size_t)B * S * D;   // 8,388,608 elements
  const size_t WS = (size_t)D * D;       // 1,048,576 elements

  char* p = (char*)d_ws;
  __bf16* xq_b  = (__bf16*)p; p += MS * 2;
  __bf16* xkv_b = (__bf16*)p; p += MS * 2;
  __bf16* Wq_b  = (__bf16*)p; p += WS * 2;
  __bf16* Wk_b  = (__bf16*)p; p += WS * 2;
  __bf16* Wv_b  = (__bf16*)p; p += WS * 2;
  __bf16* Wo_b  = (__bf16*)p; p += WS * 2;
  __bf16* q_b   = (__bf16*)p; p += MS * 2;
  __bf16* k_b   = (__bf16*)p; p += MS * 2;
  __bf16* v_b   = (__bf16*)p; p += MS * 2;
  __bf16* ctx_b = (__bf16*)p; p += MS * 2;

  // 1) precision conversion
  cvt_f32_bf16<<<4096, 256, 0, stream>>>(x_q,  xq_b,  (int)MS);
  cvt_f32_bf16<<<4096, 256, 0, stream>>>(x_kv, xkv_b, (int)MS);
  cvt_f32_bf16<<<1024, 256, 0, stream>>>(W_q, Wq_b, (int)WS);
  cvt_f32_bf16<<<1024, 256, 0, stream>>>(W_k, Wk_b, (int)WS);
  cvt_f32_bf16<<<1024, 256, 0, stream>>>(W_v, Wv_b, (int)WS);
  cvt_f32_bf16<<<1024, 256, 0, stream>>>(W_o, Wo_b, (int)WS);

  // 2) QKV projections (M=8192, N=1024, K=1024)
  dim3 gproj(8192 / 128, 1024 / 128);
  gemm_nt_wmma<__bf16><<<gproj, 256, 0, stream>>>(xq_b,  Wq_b, b_q, q_b, 8192, 1024, 1024);
  gemm_nt_wmma<__bf16><<<gproj, 256, 0, stream>>>(xkv_b, Wk_b, b_k, k_b, 8192, 1024, 1024);
  gemm_nt_wmma<__bf16><<<gproj, 256, 0, stream>>>(xkv_b, Wv_b, b_v, v_b, 8192, 1024, 1024);

  // 3) flash attention (32 q-tiles x 64 batch*head)
  flash_attn_wmma<<<dim3(S / 64, B * H), 256, 0, stream>>>(q_b, k_b, v_b, ctx_b, S, D, H);

  // 4) output projection -> f32 d_out
  gemm_nt_wmma<float><<<gproj, 256, 0, stream>>>(ctx_b, Wo_b, b_o, out, 8192, 1024, 1024);
}
